// MultiHeadAttention_82987358093746
// MI455X (gfx1250) — matrix operand order repaired and checked by reading, not compiled, not run on hardware
//
#include <hip/hip_runtime.h>

// ---------------------------------------------------------------------------
// MHA forward for MI455X (gfx1250, wave32, WMMA bf16 -> f32 accumulate)
//   B=2, S=2048, E=1024, H=16, Dh=64
//   d_out = [ out (B*S*E f32) | attn (B*H*S*S f32) ]
// Dual-accumulator K-loops: consecutive v_wmma ops are independent, avoiding
// the 5-NOP WMMA->WMMA RAW hazard (ISA 7.12.1) and keeping the XDL pipe fed.
// ---------------------------------------------------------------------------

typedef __bf16 bf16_t;
typedef __attribute__((ext_vector_type(16))) __bf16 v16bf;
typedef __attribute__((ext_vector_type(8)))  __bf16 v8bf;
typedef __attribute__((ext_vector_type(8)))  float  v8f;

#define EMBED 1024
#define HEADS 16
#define HDIM  64
#define BATCH 2
#define SEQ   2048
#define MROWS (BATCH * SEQ)   // 4096
#define ATTN_SCALE 0.125f     // 1/sqrt(64)

static __device__ __forceinline__ v8f wmma_bf16(v16bf a, v16bf b, v8f c) {
  // (neg_a, A, neg_b, B, c_mod, C, reuse_a, reuse_b)
  return __builtin_amdgcn_wmma_f32_16x16x32_bf16(false, a, false, b, (short)0, c,
                                                 false, false);
}

// A-fragment (16x32, M x K): lane holds row m = lane&15; half = lane>>4.
// elements 0..15 = K[half*16 .. half*16+15] (contiguous, matches B).
static __device__ __forceinline__ v16bf a_frag_f32(const float* __restrict__ p, int hf) {
  v16bf a;
#pragma unroll
  for (int i = 0; i < 8; ++i) a[i]     = (__bf16)p[hf * 16 + i];
#pragma unroll
  for (int i = 0; i < 8; ++i) a[8 + i] = (__bf16)p[hf * 16 + 8 + i];
  return a;
}
static __device__ __forceinline__ v16bf a_frag_bf16(const bf16_t* p, int hf) {
  v8bf r0 = *(const v8bf*)(p + hf * 16);
  v8bf r1 = *(const v8bf*)(p + hf * 16 + 8);
  v16bf a;
#pragma unroll
  for (int i = 0; i < 8; ++i) { a[i] = r0[i]; a[8 + i] = r1[i]; }
  return a;
}
// B-fragment (32x16, K x N): lane holds col n = lane&15; half = lane>>4 picks
// K range half*16..half*16+15 -> 16 contiguous values (caller folds half*16).
static __device__ __forceinline__ v16bf b_frag_f32(const float* __restrict__ p) {
  v16bf b;
#pragma unroll
  for (int i = 0; i < 16; ++i) b[i] = (__bf16)p[i];
  return b;
}
static __device__ __forceinline__ v16bf b_frag_bf16(const bf16_t* p) {
  v8bf r0 = *(const v8bf*)p;
  v8bf r1 = *(const v8bf*)(p + 8);
  v16bf b;
#pragma unroll
  for (int i = 0; i < 8; ++i) { b[i] = r0[i]; b[8 + i] = r1[i]; }
  return b;
}

// ---------------------------------------------------------------------------
// Projection: Out = X @ W^T + bias,  X:[4096,1024] f32, W:[1024,1024] f32.
// mode 0: Out bf16 laid out [B,H,S,Dh]   (for Q and K)
// mode 1: Out bf16 laid out [B,H,Dh,S]   (transposed, for V)
// grid (M/16, N/64), block 128 (4 waves); wave w owns 16x16 tile at col w*16.
// ---------------------------------------------------------------------------
__global__ void mha_proj_kernel(const float* __restrict__ X,
                                const float* __restrict__ W,
                                const float* __restrict__ bias,
                                bf16_t* __restrict__ Out, int mode) {
  const int lane = threadIdx.x & 31;
  const int hf   = lane >> 4;
  const int wv   = threadIdx.x >> 5;
  const int mBase = blockIdx.x * 16;
  const int nBase = blockIdx.y * 64 + wv * 16;
  const int mA = mBase + (lane & 15);
  const int nB = nBase + (lane & 15);

  const float* __restrict__ arow = X + (size_t)mA * EMBED;
  const float* __restrict__ brow = W + (size_t)nB * EMBED;

  v8f acc0 = {}, acc1 = {};
#pragma unroll 2
  for (int kb = 0; kb < EMBED; kb += 64) {
    if ((kb & 255) == 0) __builtin_prefetch(brow + kb + 256, 0, 0);
    v16bf a0 = a_frag_f32(arow + kb, hf);
    v16bf b0 = b_frag_f32(brow + kb + hf * 16);
    v16bf a1 = a_frag_f32(arow + kb + 32, hf);
    v16bf b1 = b_frag_f32(brow + kb + 32 + hf * 16);
    acc0 = wmma_bf16(a0, b0, acc0);   // independent of acc1 chain
    acc1 = wmma_bf16(a1, b1, acc1);
  }

  const float bn = bias[nB];
#pragma unroll
  for (int r = 0; r < 8; ++r) {
    int m = mBase + r + hf * 8;
    int b = m >> 11, s = m & (SEQ - 1);
    int h = nB >> 6, dh = nB & (HDIM - 1);
    float val = acc0[r] + acc1[r] + bn;
    size_t idx;
    if (mode == 0) idx = ((size_t)((b * HEADS + h) * SEQ + s)) * HDIM + dh;
    else           idx = ((size_t)((b * HEADS + h) * HDIM + dh)) * SEQ + s;
    Out[idx] = (__bf16)val;
  }
}

// ---------------------------------------------------------------------------
// Fused attention: one workgroup = one (b,h) x 16 query rows.
//   phase 1: scores S[16,2048] = scale * Q Kt       (8 waves, WMMA)
//   phase 2: row softmax in LDS, write attn (f32) + P (bf16) once
//   phase 3: attended[16,64] = P @ V               (4 waves, WMMA, A from LDS)
// grid (S/16, B*H), block 256, dyn-LDS ~193 KB (CDNA5: 320 KB/WGP).
// ---------------------------------------------------------------------------
__global__ void mha_attn_kernel(const bf16_t* __restrict__ Q,
                                const bf16_t* __restrict__ K,
                                const bf16_t* __restrict__ Vt,
                                float* __restrict__ attnOut,
                                bf16_t* __restrict__ attended) {
  extern __shared__ char smem[];
  float*  sc   = (float*)smem;                                 // 16*2048 f32
  bf16_t* pbuf = (bf16_t*)(smem + 16 * SEQ * 4);               // 16*2048 bf16
  float*  red  = (float*)(smem + 16 * SEQ * 4 + 16 * SEQ * 2); // 256 f32
  float*  rmax = red + 256;                                    // 16
  float*  rsum = rmax + 16;                                    // 16

  const int tid  = threadIdx.x;
  const int lane = tid & 31;
  const int hf   = lane >> 4;
  const int wv   = tid >> 5;
  const int bh    = blockIdx.y;            // b*HEADS + h
  const int qBase = blockIdx.x * 16;

  // ---- phase 1: scores ----
  const bf16_t* qrow = Q + ((size_t)bh * SEQ + qBase + (lane & 15)) * HDIM;
  v16bf a0 = a_frag_bf16(qrow, hf);        // dh 0..31
  v16bf a1 = a_frag_bf16(qrow + 32, hf);   // dh 32..63

  for (int kt = wv * 16; kt < wv * 16 + 16; ++kt) {
    const int keyBase = kt * 16;
    const bf16_t* krow = K + ((size_t)bh * SEQ + keyBase + (lane & 15)) * HDIM;
    v16bf b0 = b_frag_bf16(krow + hf * 16);
    v16bf b1 = b_frag_bf16(krow + 32 + hf * 16);
    v8f c0 = {}, c1 = {};
    c0 = wmma_bf16(a0, b0, c0);            // independent pair, no RAW chain
    c1 = wmma_bf16(a1, b1, c1);
#pragma unroll
    for (int r = 0; r < 8; ++r)
      sc[(r + hf * 8) * SEQ + keyBase + (lane & 15)] =
          (c0[r] + c1[r]) * ATTN_SCALE;
  }
  __syncthreads();

  // ---- phase 2: row softmax (16 threads per row, 128 cols each) ----
  const int row = tid >> 4, cc = tid & 15;
  float* rp = sc + row * SEQ + cc * 128;
  float mx = -3.4e38f;
#pragma unroll 8
  for (int i = 0; i < 128; ++i) mx = fmaxf(mx, rp[i]);
  red[row * 16 + cc] = mx;
  __syncthreads();
  if (cc == 0) {
    float m = red[row * 16];
#pragma unroll
    for (int i = 1; i < 16; ++i) m = fmaxf(m, red[row * 16 + i]);
    rmax[row] = m;
  }
  __syncthreads();
  const float m = rmax[row];
  float s = 0.f;
#pragma unroll 8
  for (int i = 0; i < 128; ++i) {
    float p = __expf(rp[i] - m);
    rp[i] = p;
    s += p;
  }
  red[row * 16 + cc] = s;
  __syncthreads();
  if (cc == 0) {
    float t = 0.f;
#pragma unroll
    for (int i = 0; i < 16; ++i) t += red[row * 16 + i];
    rsum[row] = t;
  }
  __syncthreads();

  // coalesced single write of attn + bf16 P for the PV GEMM
  float* aOut = attnOut + ((size_t)bh * SEQ + qBase) * SEQ;
  for (int idx = tid; idx < 16 * SEQ; idx += 256) {
    int r2 = idx >> 11;
    float p = sc[idx] * __frcp_rn(rsum[r2]);
    aOut[(size_t)idx] = p;                 // idx == r2*SEQ + col
    pbuf[idx] = (__bf16)p;
  }
  __syncthreads();

  // ---- phase 3: attended = P @ V (4 waves, one 16x16 dh-tile each) ----
  if (wv < 4) {
    const int dhBase = wv * 16;
    const bf16_t* vrow = Vt + ((size_t)bh * HDIM + dhBase + (lane & 15)) * SEQ;
    const bf16_t* prow = pbuf + (lane & 15) * SEQ;
    v8f acc0 = {}, acc1 = {};
#pragma unroll 2
    for (int kb = 0; kb < SEQ; kb += 64) {
      v16bf pa0 = a_frag_bf16(prow + kb, hf);
      v16bf vb0 = b_frag_bf16(vrow + kb + hf * 16);
      v16bf pa1 = a_frag_bf16(prow + kb + 32, hf);
      v16bf vb1 = b_frag_bf16(vrow + kb + 32 + hf * 16);
      acc0 = wmma_bf16(pa0, vb0, acc0);
      acc1 = wmma_bf16(pa1, vb1, acc1);
    }
    const int b = bh >> 4, h = bh & (HEADS - 1);
    const int col = h * HDIM + dhBase + (lane & 15);
#pragma unroll
    for (int r = 0; r < 8; ++r) {
      int gm = b * SEQ + qBase + r + hf * 8;
      attended[(size_t)gm * EMBED + col] = (__bf16)(acc0[r] + acc1[r]);
    }
  }
}

// ---------------------------------------------------------------------------
// Output projection: out = attended @ Wo^T + bo   (bf16 A x f32 W -> f32)
// grid (M/16, N/64), block 128.
// ---------------------------------------------------------------------------
__global__ void mha_outproj_kernel(const bf16_t* __restrict__ A,
                                   const float* __restrict__ Wo,
                                   const float* __restrict__ bo,
                                   float* __restrict__ out) {
  const int lane = threadIdx.x & 31;
  const int hf   = lane >> 4;
  const int wv   = threadIdx.x >> 5;
  const int mBase = blockIdx.x * 16;
  const int nBase = blockIdx.y * 64 + wv * 16;
  const int nB = nBase + (lane & 15);

  const bf16_t* __restrict__ arow = A  + (size_t)(mBase + (lane & 15)) * EMBED;
  const float*  __restrict__ brow = Wo + (size_t)nB * EMBED;

  v8f acc0 = {}, acc1 = {};
#pragma unroll 2
  for (int kb = 0; kb < EMBED; kb += 64) {
    if ((kb & 255) == 0) __builtin_prefetch(brow + kb + 256, 0, 0);
    v16bf a0 = a_frag_bf16(arow + kb, hf);
    v16bf b0 = b_frag_f32(brow + kb + hf * 16);
    v16bf a1 = a_frag_bf16(arow + kb + 32, hf);
    v16bf b1 = b_frag_f32(brow + kb + 32 + hf * 16);
    acc0 = wmma_bf16(a0, b0, acc0);
    acc1 = wmma_bf16(a1, b1, acc1);
  }
  const float bn = bo[nB];
#pragma unroll
  for (int r = 0; r < 8; ++r)
    out[(size_t)(mBase + r + hf * 8) * EMBED + nB] = acc0[r] + acc1[r] + bn;
}

// ---------------------------------------------------------------------------
extern "C" void kernel_launch(void* const* d_in, const int* in_sizes, int n_in,
                              void* d_out, int out_size, void* d_ws, size_t ws_size,
                              hipStream_t stream) {
  const float* query = (const float*)d_in[0];
  const float* key   = (const float*)d_in[1];
  const float* value = (const float*)d_in[2];
  const float* Wq = (const float*)d_in[3];
  const float* bq = (const float*)d_in[4];
  const float* Wk = (const float*)d_in[5];
  const float* bk = (const float*)d_in[6];
  const float* Wv = (const float*)d_in[7];
  const float* bv = (const float*)d_in[8];
  const float* Wo = (const float*)d_in[9];
  const float* bo = (const float*)d_in[10];

  const size_t PROJ_ELEMS = (size_t)MROWS * EMBED;  // 4096*1024
  bf16_t* qb   = (bf16_t*)d_ws;                     // [B,H,S,Dh]
  bf16_t* kb   = qb + PROJ_ELEMS;                   // [B,H,S,Dh]
  bf16_t* vt   = kb + PROJ_ELEMS;                   // [B,H,Dh,S]
  bf16_t* attd = vt + PROJ_ELEMS;                   // [B*S, E]

  float* out  = (float*)d_out;                      // [B,S,E]
  float* attn = out + PROJ_ELEMS;                   // [B,H,S,S]

  dim3 gProj(MROWS / 16, EMBED / 64), bProj(128);
  mha_proj_kernel<<<gProj, bProj, 0, stream>>>(query, Wq, bq, qb, 0);
  mha_proj_kernel<<<gProj, bProj, 0, stream>>>(key,   Wk, bk, kb, 0);
  mha_proj_kernel<<<gProj, bProj, 0, stream>>>(value, Wv, bv, vt, 1);

  const size_t shmem = (size_t)16 * SEQ * 4   // scores f32
                     + (size_t)16 * SEQ * 2   // P bf16
                     + 256 * 4 + 32 * 4;      // reduction scratch
  mha_attn_kernel<<<dim3(SEQ / 16, BATCH * HEADS), 256, shmem, stream>>>(
      qb, kb, vt, attn, attd);

  mha_outproj_kernel<<<gProj, bProj, 0, stream>>>(attd, Wo, bo, out);
}
